// BloomAttention_54563264529077
// MI455X (gfx1250) — compile-verified, matrix operand order checked by reading, not run on hardware
//
#include <hip/hip_runtime.h>
#include <cstdint>
#include <cstddef>

typedef __attribute__((ext_vector_type(16))) _Float16 v16h;
typedef __attribute__((ext_vector_type(8)))  _Float16 v8h;
typedef __attribute__((ext_vector_type(4)))  _Float16 v4h;
typedef __attribute__((ext_vector_type(8)))  float    v8f;
typedef __attribute__((ext_vector_type(4)))  unsigned int u32x4;
typedef __attribute__((ext_vector_type(8)))  int i32x8;
typedef __attribute__((ext_vector_type(4)))  int i32x4;

#define SEQ   2048
#define HID   2048
#define NHEAD 16
#define HDIM  128
#define BATCH 2
#define MROWS 4096   // BATCH*SEQ
#define N_QKV 6144   // 3*HID

#if defined(__gfx1250__) && __has_builtin(__builtin_amdgcn_tensor_load_to_lds)
#define USE_TDM 1
#else
#define USE_TDM 0
#endif

__device__ __forceinline__ v8f vzero8() {
  v8f z;
#pragma unroll
  for (int i = 0; i < 8; ++i) z[i] = 0.0f;
  return z;
}

// Load one 16x32 f16 WMMA operand fragment from an LDS row (row-major, K contiguous).
// Per ISA 7.12.2: lanes 0-15 hold K = {0..7, 16..23}, lanes 16-31 hold K = {8..15, 24..31}.
__device__ __forceinline__ v16h load_frag(const _Float16* rowp, int half) {
  const v8h a = *(const v8h*)(rowp + 8 * half);
  const v8h b = *(const v8h*)(rowp + 16 + 8 * half);
  v16h r;
#pragma unroll
  for (int i = 0; i < 8; ++i) { r[i] = a[i]; r[i + 8] = b[i]; }
  return r;
}

__device__ __forceinline__ v8f wmma_f16(v16h a, v16h b, v8f c) {
  return __builtin_amdgcn_wmma_f32_16x16x32_f16(false, a, false, b, (short)0, c, false, false);
}

#if USE_TDM
__device__ __forceinline__ unsigned lds_addr_of(const void* p) {
  // generic -> LDS addrspace cast; low 32 bits are the LDS byte offset
  return (unsigned)(unsigned long long)(const __attribute__((address_space(3))) char*)p;
}

// TDM: load a [rows x 128] f16 tile whose source rows are strided by 4096
// elements (Megatron-flat Q/K/V: row t*32+j), into LDS rows padded to 136 f16.
// pad_interval code 5 = 64 DWORDs (=128 f16), pad_amount code 3 = 4 DWORDs (=8 f16).
__device__ __forceinline__ void tdm_load_tile_f16(unsigned lds_off,
                                                  const _Float16* gptr, int rows) {
  unsigned long long ga = (unsigned long long)(uintptr_t)gptr;
  unsigned galo = (unsigned)__builtin_amdgcn_readfirstlane((int)(unsigned)(ga & 0xFFFFFFFFu));
  unsigned gahi = (unsigned)__builtin_amdgcn_readfirstlane((int)(unsigned)(ga >> 32));
  unsigned ldsa = (unsigned)__builtin_amdgcn_readfirstlane((int)lds_off);
  u32x4 g0;
  g0[0] = 1u;                                        // count=1 valid descriptor
  g0[1] = ldsa;                                      // lds_addr
  g0[2] = galo;                                      // global_addr[31:0]
  g0[3] = (gahi & 0x1FFFFFFu) | (2u << 30);          // global_addr[56:32] | type=2
  i32x8 g1;
  g1[0] = (int)((1u << 16) | (1u << 20) | (5u << 22) | (3u << 25)); // 2B, pad on
  g1[1] = (int)(128u << 16);                         // tensor_dim0 = 128 (lo16)
  g1[2] = 0;                                         // dim0 hi / dim1 lo (65536)
  g1[3] = (int)(1u | (128u << 16));                  // dim1 hi=1, tile_dim0=128
  g1[4] = rows;                                      // tile_dim1 = rows
  g1[5] = 4096;                                      // tensor_dim0_stride = 32*128
  g1[6] = 0;
  g1[7] = 0;
  i32x4 z4 = {0, 0, 0, 0};
  i32x8 z8 = {0, 0, 0, 0, 0, 0, 0, 0};
  __builtin_amdgcn_tensor_load_to_lds(g0, g1, z4, z4, z8, 0);
}
#endif

// ---------------------------------------------------------------------------
// GEMM: C[M,N] = A[M,K] (f32, row-major) * Bw[N,K]^T (f32, row-major) + bias.
// f32 -> f16 on the fly into LDS; f32 WMMA accumulation.
// MODE 0: QKV epilogue -> scatter f16 into Q/K/V flat [MROWS*16, 128] buffers.
// MODE 1: dense epilogue -> f32 out with [s,b,h] -> [b,s,h] permutation.
// ---------------------------------------------------------------------------
template <int MODE>
__global__ __launch_bounds__(256) void gemm_kernel(
    const float* __restrict__ A, const float* __restrict__ Bw,
    const float* __restrict__ bias,
    _Float16* __restrict__ Qo, _Float16* __restrict__ Ko, _Float16* __restrict__ Vo,
    float* __restrict__ Out, int K)
{
  __shared__ __align__(16) _Float16 As[128][40];
  __shared__ __align__(16) _Float16 Bs[128][40];

  const int tid  = threadIdx.x;
  const int lane = tid & 31;
  const int w    = tid >> 5;
  const int wm   = w & 3;
  const int wn   = w >> 2;
  const int half = lane >> 4;
  const int lr   = lane & 15;
  const int bm   = blockIdx.y;
  const int bn   = blockIdx.x;

  v8f acc[2][4];
#pragma unroll
  for (int mt = 0; mt < 2; ++mt)
#pragma unroll
    for (int nt = 0; nt < 4; ++nt) acc[mt][nt] = vzero8();

  for (int kt = 0; kt < K; kt += 32) {
#pragma unroll
    for (int i = 0; i < 4; ++i) {
      int idx = tid + 256 * i;          // 0..1023 -> 128 rows x 8 groups of 4
      int row = idx >> 3;
      int cg  = idx & 7;
      float4 va = *(const float4*)&A [(size_t)(bm * 128 + row) * K + kt + cg * 4];
      float4 vb = *(const float4*)&Bw[(size_t)(bn * 128 + row) * K + kt + cg * 4];
      v4h ha = { (_Float16)va.x, (_Float16)va.y, (_Float16)va.z, (_Float16)va.w };
      v4h hb = { (_Float16)vb.x, (_Float16)vb.y, (_Float16)vb.z, (_Float16)vb.w };
      *(v4h*)&As[row][cg * 4] = ha;
      *(v4h*)&Bs[row][cg * 4] = hb;
    }
    __syncthreads();

    v16h af[2], bf[4];
#pragma unroll
    for (int mt = 0; mt < 2; ++mt) af[mt] = load_frag(&As[wm * 32 + mt * 16 + lr][0], half);
#pragma unroll
    for (int nt = 0; nt < 4; ++nt) bf[nt] = load_frag(&Bs[wn * 64 + nt * 16 + lr][0], half);
#pragma unroll
    for (int mt = 0; mt < 2; ++mt)
#pragma unroll
      for (int nt = 0; nt < 4; ++nt)
        acc[mt][nt] = wmma_f16(af[mt], bf[nt], acc[mt][nt]);
    __syncthreads();
  }

  // Epilogue. C layout (16x16 f32 tile): VGPR slot r, lane l -> row = r + 8*(l/16), col = l%16.
#pragma unroll
  for (int mt = 0; mt < 2; ++mt) {
#pragma unroll
    for (int nt = 0; nt < 4; ++nt) {
      int coln = bn * 128 + wn * 64 + nt * 16 + lr;
      float bv = bias[coln];
#pragma unroll
      for (int r = 0; r < 8; ++r) {
        int rowm = bm * 128 + wm * 32 + mt * 16 + r + 8 * half;
        float v = acc[mt][nt][r] + bv;
        if (MODE == 0) {
          int nh  = coln / 384;
          int r3  = coln % 384;
          int sel = r3 >> 7;
          int d   = r3 & 127;
          size_t o = ((size_t)rowm * NHEAD + nh) * HDIM + d;
          _Float16* dst = (sel == 0) ? Qo : (sel == 1) ? Ko : Vo;
          dst[o] = (_Float16)v;
        } else {
          Out[(size_t)(rowm & 1) * SEQ * HID + (size_t)(rowm >> 1) * HID + coln] = v;
        }
      }
    }
  }
}

// ---------------------------------------------------------------------------
// Flash attention with online softmax. One block = 128 query rows of one
// reinterpreted batch-head j (0..31); Q/K/V flat rows index as (t*32 + j).
// Q and K tiles staged by the Tensor Data Mover; V vector-loaded + transposed.
// ---------------------------------------------------------------------------
__global__ __launch_bounds__(256) void attn_kernel(
    const _Float16* __restrict__ Qg, const _Float16* __restrict__ Kg,
    const _Float16* __restrict__ Vg, const float* __restrict__ alibi,
    const int* __restrict__ amask, float* __restrict__ Ctx)
{
  __shared__ __align__(16) _Float16 Qs[128][HDIM + 8];   // 34.8 KB
  __shared__ __align__(16) _Float16 Ks[64][HDIM + 8];    // 17.4 KB
  __shared__ __align__(16) _Float16 Vt[HDIM][64 + 8];    // 18.4 KB (V transposed)
  __shared__ __align__(16) _Float16 Ps[8][16][64 + 8];   // 18.4 KB (per-wave P staging)

  const int tid  = threadIdx.x;
  const int lane = tid & 31;
  const int w    = tid >> 5;
  const int half = lane >> 4;
  const int lr   = lane & 15;
  const int qb   = blockIdx.x * 128;
  const int j    = blockIdx.y;      // 0..31
  const int jh   = j & 15;          // alibi head index
  const int b2   = j >> 4;          // batch index after reshape(b, NH, ...)
  const float invs = 0.08838834764831845f; // 1/sqrt(128) == SCALE/NORM_FACTOR

  // ---- stage Q tile ----
#if USE_TDM
  if (w == 0) {
    tdm_load_tile_f16(lds_addr_of(&Qs[0][0]),
                      Qg + ((size_t)qb * 32 + j) * HDIM, 128);
    __builtin_amdgcn_s_wait_tensorcnt(0);
  }
#else
  {
    int row = tid >> 1;
    int hh  = tid & 1;
    const _Float16* src = Qg + ((size_t)(qb + row) * 32 + j) * HDIM + hh * 64;
#pragma unroll
    for (int c = 0; c < 8; ++c)
      *(v8h*)&Qs[row][hh * 64 + c * 8] = *(const v8h*)&src[c * 8];
  }
#endif
  __syncthreads();

  v16h qf[4];
#pragma unroll
  for (int kc = 0; kc < 4; ++kc) qf[kc] = load_frag(&Qs[w * 16 + lr][kc * 32], half);

  v8f cacc[8];
#pragma unroll
  for (int nt = 0; nt < 8; ++nt) cacc[nt] = vzero8();
  float mrow[8], lsum[8];
  bool rowpad[8];
#pragma unroll
  for (int r = 0; r < 8; ++r) {
    mrow[r] = -1e30f;
    lsum[r] = 0.0f;
    rowpad[r] = (amask[b2 * SEQ + qb + w * 16 + r + 8 * half] == 0);
  }

  const int nkt = qb / 64 + 2;     // causal bound: k-cols up to qb+127
  for (int kt = 0; kt < nkt; ++kt) {
    const int kb = kt * 64;
    __syncthreads();               // protect Ks/Vt reuse across iterations

    // ---- stage K (TDM) while all waves stage V (vector + transpose) ----
#if USE_TDM
    if (w == 0)
      tdm_load_tile_f16(lds_addr_of(&Ks[0][0]),
                        Kg + ((size_t)kb * 32 + j) * HDIM, 64);
#endif
#pragma unroll
    for (int i = 0; i < 4; ++i) {
      int idx = tid + 256 * i;     // 64 rows x 16 col-groups
      int row = idx >> 4;
      int cg  = idx & 15;
#if !USE_TDM
      *(v8h*)&Ks[row][cg * 8] = *(const v8h*)&Kg[((size_t)(kb + row) * 32 + j) * HDIM + cg * 8];
#endif
      v8h vv = *(const v8h*)&Vg[((size_t)(kb + row) * 32 + j) * HDIM + cg * 8];
#pragma unroll
      for (int e = 0; e < 8; ++e) Vt[cg * 8 + e][row] = vv[e];
      if (kt + 1 < nkt)
        __builtin_prefetch(&Vg[((size_t)(kb + 64 + row) * 32 + j) * HDIM + cg * 8], 0, 1);
    }
#if USE_TDM
    if (w == 0) __builtin_amdgcn_s_wait_tensorcnt(0);
#endif
    __syncthreads();

    // ---- scores 16x64 per wave: S = Q K^T (batched frag loads, then WMMAs) ----
    v8f sacc[4];
#pragma unroll
    for (int nt = 0; nt < 4; ++nt) sacc[nt] = vzero8();
#pragma unroll
    for (int kc = 0; kc < 4; ++kc) {
      v16h bf[4];
#pragma unroll
      for (int nt = 0; nt < 4; ++nt) bf[nt] = load_frag(&Ks[nt * 16 + lr][kc * 32], half);
#pragma unroll
      for (int nt = 0; nt < 4; ++nt) sacc[nt] = wmma_f16(qf[kc], bf[nt], sacc[nt]);
    }

    // ---- scale + alibi + causal/pad mask ----
#pragma unroll
    for (int nt = 0; nt < 4; ++nt) {
      int col = kb + nt * 16 + lr;
      float al = alibi[jh * SEQ + col];
#pragma unroll
      for (int r = 0; r < 8; ++r) {
        int qrow = qb + w * 16 + r + 8 * half;
        float s = sacc[nt][r] * invs + al;
        if (col > qrow || rowpad[r]) s = -10000.0f;
        sacc[nt][r] = s;
      }
    }

    // ---- online softmax (row stats live in 16-lane halves; wave32 shuffles) ----
#pragma unroll
    for (int r = 0; r < 8; ++r) {
      float mx = sacc[0][r];
#pragma unroll
      for (int nt = 1; nt < 4; ++nt) mx = fmaxf(mx, sacc[nt][r]);
#pragma unroll
      for (int xm = 1; xm < 16; xm <<= 1) mx = fmaxf(mx, __shfl_xor(mx, xm, 32));
      float mnew = fmaxf(mrow[r], mx);
      float scl  = __expf(mrow[r] - mnew);
      mrow[r] = mnew;
      float rs = 0.0f;
#pragma unroll
      for (int nt = 0; nt < 4; ++nt) {
        float p = __expf(sacc[nt][r] - mnew);
        rs += p;
        Ps[w][r + 8 * half][nt * 16 + lr] = (_Float16)p;   // C-layout -> row-major
      }
#pragma unroll
      for (int xm = 1; xm < 16; xm <<= 1) rs += __shfl_xor(rs, xm, 32);
      lsum[r] = lsum[r] * scl + rs;
#pragma unroll
      for (int nt = 0; nt < 8; ++nt) cacc[nt][r] *= scl;
    }

    // ---- ctx += P @ V (per-wave P from LDS; DS is in-order within a wave) ----
    v16h pf[2];
    pf[0] = load_frag(&Ps[w][lr][0], half);
    pf[1] = load_frag(&Ps[w][lr][32], half);
#pragma unroll
    for (int kc = 0; kc < 2; ++kc) {
#pragma unroll
      for (int g = 0; g < 2; ++g) {
        v16h bf[4];
#pragma unroll
        for (int q = 0; q < 4; ++q)
          bf[q] = load_frag(&Vt[(g * 4 + q) * 16 + lr][kc * 32], half);
#pragma unroll
        for (int q = 0; q < 4; ++q)
          cacc[g * 4 + q] = wmma_f16(pf[kc], bf[q], cacc[g * 4 + q]);
      }
    }
  }

  // ---- normalize, scatter ctx as [s, b, h]: row = t*2 + b2, col = (j%16)*128 + d
#pragma unroll
  for (int nt = 0; nt < 8; ++nt) {
    int col = jh * HDIM + nt * 16 + lr;
#pragma unroll
    for (int r = 0; r < 8; ++r) {
      int t = qb + w * 16 + r + 8 * half;
      Ctx[(size_t)(t * 2 + b2) * HID + col] = cacc[nt][r] / lsum[r];
    }
  }
}

// ---------------------------------------------------------------------------
extern "C" void kernel_launch(void* const* d_in, const int* in_sizes, int n_in,
                              void* d_out, int out_size, void* d_ws, size_t ws_size,
                              hipStream_t stream) {
  const float* hs    = (const float*)d_in[0];  // [B,S,H] f32
  const float* alibi = (const float*)d_in[1];  // [NH,1,S] f32
  const float* wqkv  = (const float*)d_in[2];  // [3H,H] f32
  const float* bqkv  = (const float*)d_in[3];  // [3H] f32
  const float* wdn   = (const float*)d_in[4];  // [H,H] f32
  const float* bdn   = (const float*)d_in[5];  // [H] f32
  const int*   amask = (const int*)d_in[6];    // [B,S] i32
  float* out = (float*)d_out;                  // [B,S,H] f32
  (void)in_sizes; (void)n_in; (void)out_size; (void)ws_size;

  char* ws = (char*)d_ws;
  size_t off = 0;
  auto take = [&](size_t bytes) -> void* {
    void* p = ws + off;
    off += (bytes + 255) & ~(size_t)255;
    return p;
  };
  _Float16* Qb  = (_Float16*)take((size_t)MROWS * HID * sizeof(_Float16)); // 16 MB
  _Float16* Kb  = (_Float16*)take((size_t)MROWS * HID * sizeof(_Float16)); // 16 MB
  _Float16* Vb  = (_Float16*)take((size_t)MROWS * HID * sizeof(_Float16)); // 16 MB
  float*    Ctx = (float*)take((size_t)MROWS * HID * sizeof(float));       // 32 MB

  // 1) QKV projection: [4096,2048] x [6144,2048]^T + bias -> Q/K/V f16 flat
  gemm_kernel<0><<<dim3(N_QKV / 128, MROWS / 128), 256, 0, stream>>>(
      hs, wqkv, bqkv, Qb, Kb, Vb, nullptr, HID);

  // 2) fused scaled-masked-softmax attention, 32 batch-heads x 16 q-tiles
  attn_kernel<<<dim3(SEQ / 128, BATCH * NHEAD), 256, 0, stream>>>(
      Qb, Kb, Vb, alibi, amask, Ctx);

  // 3) dense projection: ctx[s,b,h] x [2048,2048]^T + bias -> out[b,s,h]
  gemm_kernel<1><<<dim3(HID / 128, MROWS / 128), 256, 0, stream>>>(
      Ctx, wdn, bdn, nullptr, nullptr, nullptr, out, HID);
}
